// MultilevelEncoder_18098992185623
// MI455X (gfx1250) — matrix operand order repaired and checked by reading, not compiled, not run on hardware
//
#include <hip/hip_runtime.h>
#include <hip/hip_bf16.h>
#include <stdint.h>

// ---------------- problem constants ----------------
#define BB   128
#define LL   64
#define FF   3072
#define DD   1024
#define CCV  512
#define CCN  1024
#define MM   (BB * LL)        // 8192 rows
#define LP   (LL + 2)         // padded time dim for 3-tap conv
#define BN_EPS 1e-5f
#define KC   32               // K-chunk per WMMA step
#define LDP  40               // padded LDS row stride in elements (80 B, bank-conflict-free)

typedef __attribute__((ext_vector_type(16))) __bf16        v16bf;
typedef __attribute__((ext_vector_type(8)))  float         v8f;
typedef __attribute__((ext_vector_type(4)))  unsigned int  v4u;
typedef __attribute__((ext_vector_type(8)))  unsigned int  v8u;
typedef __attribute__((ext_vector_type(4)))  unsigned int  u32x4;
typedef __attribute__((ext_vector_type(8)))  unsigned int  u32x8;
typedef __attribute__((ext_vector_type(4)))  int           i32x4;
typedef __attribute__((ext_vector_type(8)))  int           i32x8;

// ---------------- helpers ----------------
__device__ __forceinline__ __bf16 f2bf(float x) {
    unsigned u = __builtin_bit_cast(unsigned, x);
    unsigned short h = (unsigned short)((u + 0x7FFFu + ((u >> 16) & 1u)) >> 16);
    return __builtin_bit_cast(__bf16, h);
}

// A-matrix 16x32 bf16 fragment (ISA 7.12.2):
// lanes 0-15: K = {0..7} in v0..3, {16..23} in v4..7 ; lanes 16-31: +8
__device__ __forceinline__ v16bf load_fragA(const __bf16* p, int hl) {
    const v4u lo = *(const v4u*)(p + 8 * hl);
    const v4u hi = *(const v4u*)(p + 16 + 8 * hl);
    v8u w = __builtin_shufflevector(lo, hi, 0, 1, 2, 3, 4, 5, 6, 7);
    return __builtin_bit_cast(v16bf, w);
}

// B-matrix 32x16 bf16 fragment: lanes 0-15 hold K={0..7,16..23}, lanes 16-31 +8
__device__ __forceinline__ v16bf load_fragB(const __bf16* p, int hl) {
    const __bf16* q = p + 8 * hl;
    const v4u lo = *(const v4u*)(q);
    const v4u hi = *(const v4u*)(q + 16);
    v8u w = __builtin_shufflevector(lo, hi, 0, 1, 2, 3, 4, 5, 6, 7);
    return __builtin_bit_cast(v16bf, w);
}

__device__ __forceinline__ v8f bwmma(v16bf a, v16bf b, v8f c) {
    return __builtin_amdgcn_wmma_f32_16x16x32_bf16(false, a, false, b,
                                                   (short)0, c, false, false);
}

__device__ __forceinline__ unsigned lds_off(const void* p) {
    return (unsigned)(unsigned long long)(uintptr_t)p;   // generic LDS ptr low 32 = LDS byte offset
}

// ---- Tensor Data Mover: 3D tile (tile0 elems x tile1 rows x tile2 z) -> LDS ----
// Row stride = rowStride elems (dim0_stride); z stride = zStride elems (dim1_stride).
// pad_enable pads each 64 B row chunk with 16 B in LDS -> 80 B LDS row stride (LDP).
__device__ __forceinline__ void tdm_load_tile(const void* gptr, unsigned ldsoff,
                                              unsigned dimK,
                                              unsigned tile0, unsigned tile1, unsigned tile2,
                                              unsigned long long rowStride,
                                              unsigned long long zStride) {
    unsigned long long ga = (unsigned long long)(uintptr_t)gptr;
    u32x4 g0;
    g0[0] = 1u;                                                 // count=1, normal D#
    g0[1] = ldsoff;                                             // lds_addr (bytes)
    g0[2] = (unsigned)ga;                                       // global_addr[31:0]
    g0[3] = ((unsigned)(ga >> 32) & 0x01FFFFFFu) | (2u << 30);  // addr[56:32] | type=2
    const unsigned dim1 = 1u << 20;                             // huge (no clipping)
    u32x8 g1;
    g1[0] = (1u << 16)                // data_size=1 (2 bytes/elem)
          | (1u << 20)                // pad_enable
          | (3u << 22)                // pad_interval: 16 dwords (64 B)
          | (3u << 25);               // pad_amount:   4 dwords (16 B)
    g1[1] = (dimK & 0xFFFFu) << 16;                    // tensor_dim0[15:0]
    g1[2] = (dimK >> 16) | ((dim1 & 0xFFFFu) << 16);   // dim0[31:16] | dim1[15:0]
    g1[3] = (dim1 >> 16) | (tile0 << 16);              // dim1[31:16] | tile_dim0
    g1[4] = tile1 | (tile2 << 16);                     // tile_dim1 | tile_dim2
    g1[5] = (unsigned)rowStride;                       // dim0_stride[31:0]
    g1[6] = ((unsigned)(rowStride >> 32) & 0xFFFFu)    // dim0_stride[47:32]
          | ((unsigned)(zStride & 0xFFFFu) << 16);     // dim1_stride[15:0]
    g1[7] = (unsigned)(zStride >> 16);                 // dim1_stride[47:16]
    i32x4 g2;
    g2[0] = (int)(1u << 20);                           // tensor_dim2 (huge)
    g2[1] = 0; g2[2] = 0; g2[3] = 0;                   // dim3 / dim2_stride / tile_dim3 unused
    i32x4 g3 = (i32x4)0;
#if __has_builtin(__builtin_amdgcn_tensor_load_to_lds)
#if __clang_major__ >= 23
    __builtin_amdgcn_tensor_load_to_lds(g0, __builtin_bit_cast(i32x8, g1),
                                        g2, g3, (i32x8)0, 0);
#else
    __builtin_amdgcn_tensor_load_to_lds(g0, __builtin_bit_cast(i32x8, g1),
                                        g2, g3, 0);
#endif
#else
    asm volatile("tensor_load_to_lds %0, %1, %2, %3"
                 :: "s"(g0), "s"(g1), "s"(g2), "s"(g3) : "memory");
#endif
}

// ---------------- conversion / layout kernels ----------------
__global__ void k_cvt_bf16(const float* __restrict__ in, __bf16* __restrict__ out, int n) {
    int i = (blockIdx.x * blockDim.x + threadIdx.x) * 4;
    if (i + 3 < n) {
        out[i + 0] = f2bf(in[i + 0]);
        out[i + 1] = f2bf(in[i + 1]);
        out[i + 2] = f2bf(in[i + 2]);
        out[i + 3] = f2bf(in[i + 3]);
    } else {
        for (; i < n; ++i) out[i] = f2bf(in[i]);
    }
}

// W (K x N, row-major) -> Wt (N x K, row-major, bf16)
__global__ void k_transpose_cvt(const float* __restrict__ W, __bf16* __restrict__ Wt,
                                int K, int N) {
    int i = blockIdx.x * blockDim.x + threadIdx.x;
    if (i >= K * N) return;
    int n = i / K, k = i % K;
    Wt[i] = f2bf(W[(size_t)k * N + n]);
}

// Kv (CV, D, 3) -> Kvt[t][c][k], bf16
__global__ void k_kv_transpose(const float* __restrict__ Kv, __bf16* __restrict__ Kvt) {
    int i = blockIdx.x * blockDim.x + threadIdx.x;
    if (i >= 3 * CCV * DD) return;
    int t = i / (CCV * DD);
    int rem = i % (CCV * DD);
    int c = rem / DD, k = rem % DD;
    Kvt[i] = f2bf(Kv[((size_t)c * DD + k) * 3 + t]);
}

// e1 (B, L, D) fp32 -> zero-padded (B, L+2, D) bf16
__global__ void k_pad_e1(const float* __restrict__ e1, __bf16* __restrict__ e1p) {
    int i = blockIdx.x * blockDim.x + threadIdx.x;
    if (i >= BB * LP * DD) return;
    int b = i / (LP * DD);
    int rem = i % (LP * DD);
    int l = rem / DD, d = rem % DD;
    float v = (l == 0 || l == LP - 1) ? 0.0f
                                      : e1[((size_t)b * LL + (l - 1)) * DD + d];
    e1p[i] = f2bf(v);
}

// ---------------- WMMA GEMM with TDM double-buffered LDS tiles ----------------
// C(M x N) = A(M x K) @ Bt(N x K)^T.  Block: 256 thr = 8 waves (4 M x 2 N).
// Block tile 128x128, K-chunk 32; wave tile 32x64 (8 WMMAs / chunk, 6 frag loads).
// A rows are staged as a TDM 3D tile: (32 k) x (64 rows) x (2 groups) where
// consecutive 64-row groups are aZstride elements apart (66*D for padded e1,
// 64*K for a plain dense matrix). LDS rows padded to 80 B (bank-conflict-free).
// flags: bit0 = accumulate into existing C, bit1 = batchnorm+sigmoid epilogue.
__global__ __launch_bounds__(256)
void k_gemm_wmma(const __bf16* __restrict__ A, const __bf16* __restrict__ Bt,
                 const float* __restrict__ bias,
                 const float* __restrict__ g, const float* __restrict__ beta,
                 const float* __restrict__ mean, const float* __restrict__ var,
                 float* __restrict__ C, int N, int K,
                 unsigned long long aZstride, int flags) {
    __shared__ alignas(16) __bf16 Atile[2][128 * LDP];   // 20 KB
    __shared__ alignas(16) __bf16 Btile[2][128 * LDP];   // 20 KB

    const int tid  = threadIdx.x;
    const int wave = tid >> 5;
    const int lane = tid & 31;
    const int hl = lane >> 4, lm = lane & 15;
    const int wm = wave >> 1;           // 0..3 : 32-row band
    const int wn = wave & 1;            // 0..1 : 64-col band
    const int mblk = blockIdx.x * 128;
    const int nblk = blockIdx.y * 128;

    const __bf16* Ablk = A  + (size_t)(mblk >> 6) * aZstride;
    const __bf16* Bblk = Bt + (size_t)nblk * K;
    const int ntiles = K >> 5;

    if (wave == 0) {   // TDM: stage tile 0 into buffer 0 (A,B: 32k x 64r x 2z)
        tdm_load_tile(Ablk, lds_off(Atile[0]), K, KC, 64, 2,
                      (unsigned long long)K, aZstride);
        tdm_load_tile(Bblk, lds_off(Btile[0]), K, KC, 64, 2,
                      (unsigned long long)K, 64ull * (unsigned)K);
    }

    v8f acc[2][4];
#pragma unroll
    for (int i = 0; i < 2; ++i)
#pragma unroll
        for (int j = 0; j < 4; ++j) acc[i][j] = {};

    for (int t = 0; t < ntiles; ++t) {
        const int buf = t & 1;
        if (wave == 0) {
            if (t + 1 < ntiles) {     // prefetch next K-chunk via TDM while computing
                const int kc = (t + 1) << 5;
                tdm_load_tile(Ablk + kc, lds_off(Atile[buf ^ 1]), K, KC, 64, 2,
                              (unsigned long long)K, aZstride);
                tdm_load_tile(Bblk + kc, lds_off(Btile[buf ^ 1]), K, KC, 64, 2,
                              (unsigned long long)K, 64ull * (unsigned)K);
                __builtin_amdgcn_s_wait_tensorcnt(2);   // tile t resident
            } else {
                __builtin_amdgcn_s_wait_tensorcnt(0);
            }
        }
        __syncthreads();

        const __bf16* As = Atile[buf];
        const __bf16* Bs = Btile[buf];
        v16bf a0 = load_fragA(As + (wm * 32 + lm) * LDP, hl);
        v16bf a1 = load_fragA(As + (wm * 32 + 16 + lm) * LDP, hl);
        v16bf b[4];
#pragma unroll
        for (int j = 0; j < 4; ++j)
            b[j] = load_fragB(Bs + (wn * 64 + j * 16 + lm) * LDP, hl);
#pragma unroll
        for (int j = 0; j < 4; ++j) {
            acc[0][j] = bwmma(a0, b[j], acc[0][j]);
            acc[1][j] = bwmma(a1, b[j], acc[1][j]);
        }
        __syncthreads();   // tile consumed; its buffer may be refilled next iter
    }

    // -------- epilogue --------
    int   cc[4];
    float bs[4], sc[4], mn[4], bt[4];
#pragma unroll
    for (int j = 0; j < 4; ++j) {
        const int c = nblk + wn * 64 + j * 16 + lm;
        cc[j] = c;
        bs[j] = bias ? bias[c] : 0.0f;
        sc[j] = 1.f; mn[j] = 0.f; bt[j] = 0.f;
        if (flags & 2) {
            sc[j] = g[c] * rsqrtf(var[c] + BN_EPS);
            mn[j] = mean[c];
            bt[j] = beta[c];
        }
    }
#pragma unroll
    for (int i = 0; i < 2; ++i) {
#pragma unroll
        for (int v = 0; v < 8; ++v) {
            const int r = mblk + wm * 32 + i * 16 + v + 8 * hl;
#pragma unroll
            for (int j = 0; j < 4; ++j) {
                float x = acc[i][j][v] + bs[j];
                if (flags & 1) x += C[(size_t)r * N + cc[j]];
                if (flags & 2) x = 1.0f / (1.0f + __expf(-((x - mn[j]) * sc[j] + bt[j])));
                C[(size_t)r * N + cc[j]] = x;
            }
        }
    }
}

// ---------------- masked-softmax attention pooling ----------------
__global__ void k_attn_pool(const float* __restrict__ e0, const float* __restrict__ w_attn,
                            const float* __restrict__ b_attn, const int* __restrict__ lens,
                            float* __restrict__ sent) {
    __shared__ float sc[LL];
    __shared__ float at[LL];
    const int b = blockIdx.x;
    const int tid = threadIdx.x;       // 256 threads = 8 waves
    const int wave = tid >> 5, lane = tid & 31;

    for (int l = wave; l < LL; l += 8) {
        const float* row = e0 + ((size_t)b * LL + l) * DD;
        float s = 0.f;
        for (int j = lane; j < DD; j += 32) s += row[j] * w_attn[j];
        for (int off = 16; off > 0; off >>= 1) s += __shfl_down(s, off, 32);
        if (lane == 0) sc[l] = s + b_attn[0];
    }
    __syncthreads();

    if (tid == 0) {
        int len = lens[b];
        float mx = -1e30f;
        for (int l = 0; l < LL; ++l) {
            float v = (l < len) ? sc[l] : -1e18f;
            sc[l] = v;
            mx = fmaxf(mx, v);
        }
        float sum = 0.f;
        for (int l = 0; l < LL; ++l) { float e = __expf(sc[l] - mx); at[l] = e; sum += e; }
        float inv = 1.0f / sum;
        for (int l = 0; l < LL; ++l) at[l] *= inv;
    }
    __syncthreads();

    for (int d = tid; d < DD; d += 256) {
        float s = 0.f;
        for (int l = 0; l < LL; ++l) s += at[l] * e0[((size_t)b * LL + l) * DD + d];
        sent[(size_t)b * DD + d] = s;
    }
}

// ---------------- per-(b, class) top-k mean ----------------
__global__ void k_topk_mean(const float* __restrict__ logits, const int* __restrict__ lens,
                            float* __restrict__ out, int C) {
    int idx = blockIdx.x * blockDim.x + threadIdx.x;
    if (idx >= BB * C) return;
    int b = idx / C, c = idx % C;
    int len = lens[b];
    int k = (len + 7) >> 3;
    float top[8];
#pragma unroll
    for (int i = 0; i < 8; ++i) top[i] = -1e30f;
    for (int l = 0; l < len; ++l) {
        float v = logits[((size_t)b * LL + l) * C + c];
        if (v > top[7]) {
            top[7] = v;
#pragma unroll
            for (int i = 7; i > 0; --i) {
                if (top[i] > top[i - 1]) { float t = top[i - 1]; top[i - 1] = top[i]; top[i] = t; }
            }
        }
    }
    float s = 0.f;
    for (int i = 0; i < 8; ++i) s += (i < k) ? top[i] : 0.f;
    out[idx] = s / (float)k;
}

// ---------------- host-side orchestration ----------------
extern "C" void kernel_launch(void* const* d_in, const int* in_sizes, int n_in,
                              void* d_out, int out_size, void* d_ws, size_t ws_size,
                              hipStream_t stream) {
    const float* inputs  = (const float*)d_in[0];
    const int*   lens    = (const int*)d_in[1];
    const float* W0      = (const float*)d_in[2];
    const float* b0      = (const float*)d_in[3];
    const float* W1      = (const float*)d_in[4];
    const float* b1      = (const float*)d_in[5];
    const float* W2      = (const float*)d_in[6];
    const float* b2      = (const float*)d_in[7];
    const float* w_attn  = (const float*)d_in[8];
    const float* b_attn  = (const float*)d_in[9];
    const float* Kv      = (const float*)d_in[10];
    const float* bv      = (const float*)d_in[11];
    const float* g_v     = (const float*)d_in[12];
    const float* beta_v  = (const float*)d_in[13];
    const float* mean_v  = (const float*)d_in[14];
    const float* var_v   = (const float*)d_in[15];
    const float* Kn      = (const float*)d_in[16];
    const float* bn_b    = (const float*)d_in[17];
    const float* g_n     = (const float*)d_in[18];
    const float* beta_n  = (const float*)d_in[19];
    const float* mean_n  = (const float*)d_in[20];
    const float* var_n   = (const float*)d_in[21];

    // output layout (floats): sent | e1 | e2 | inst_verb | inst_noun
    float* out     = (float*)d_out;
    float* o_sent  = out;
    float* o_e1    = out + (size_t)BB * DD;
    float* o_e2    = o_e1 + (size_t)MM * DD;
    float* o_ilv   = o_e2 + (size_t)MM * DD;
    float* o_iln   = o_ilv + (size_t)BB * CCV;

    // workspace layout
    char* ws = (char*)d_ws;
    size_t off = 0;
    __bf16* Xbf  = (__bf16*)(ws + off); off += (size_t)MM * FF * 2;
    __bf16* W0t  = (__bf16*)(ws + off); off += (size_t)DD * FF * 2;
    __bf16* W1t  = (__bf16*)(ws + off); off += (size_t)DD * FF * 2;
    __bf16* W2t  = (__bf16*)(ws + off); off += (size_t)DD * FF * 2;
    float*  e0   = (float*)(ws + off);  off += (size_t)MM * DD * 4;
    __bf16* E1p  = (__bf16*)(ws + off); off += (size_t)BB * LP * DD * 2;
    __bf16* E2b  = (__bf16*)(ws + off); off += (size_t)MM * DD * 2;
    __bf16* Kvt  = (__bf16*)(ws + off); off += (size_t)3 * CCV * DD * 2;
    __bf16* Knt  = (__bf16*)(ws + off); off += (size_t)CCN * DD * 2;
    float*  LVrb = (float*)(ws + off);  off += (size_t)MM * CCV * 4;
    float*  LNun = (float*)(ws + off);  off += (size_t)MM * CCN * 4;
    (void)off; (void)ws_size; (void)out_size; (void)n_in; (void)in_sizes;

    // 1) input + weight conversion
    {
        int n = MM * FF;
        k_cvt_bf16<<<(n / 4 + 255) / 256, 256, 0, stream>>>(inputs, Xbf, n);
        int nw = FF * DD;
        k_transpose_cvt<<<(nw + 255) / 256, 256, 0, stream>>>(W0, W0t, FF, DD);
        k_transpose_cvt<<<(nw + 255) / 256, 256, 0, stream>>>(W1, W1t, FF, DD);
        k_transpose_cvt<<<(nw + 255) / 256, 256, 0, stream>>>(W2, W2t, FF, DD);
        int nkv = 3 * CCV * DD;
        k_kv_transpose<<<(nkv + 255) / 256, 256, 0, stream>>>(Kv, Kvt);
        int nkn = CCN * DD;   // Kn is (CN, D, 1) == already N x K
        k_cvt_bf16<<<(nkn / 4 + 255) / 256, 256, 0, stream>>>(Kn, Knt, nkn);
    }

    // 2) three embedding GEMMs: (8192 x 3072) @ (3072 x 1024)
    {
        dim3 grid(MM / 128, DD / 128);
        k_gemm_wmma<<<grid, 256, 0, stream>>>(Xbf, W0t, b0, nullptr, nullptr, nullptr, nullptr,
                                              e0, DD, FF, 64ull * FF, 0);
        k_gemm_wmma<<<grid, 256, 0, stream>>>(Xbf, W1t, b1, nullptr, nullptr, nullptr, nullptr,
                                              o_e1, DD, FF, 64ull * FF, 0);
        k_gemm_wmma<<<grid, 256, 0, stream>>>(Xbf, W2t, b2, nullptr, nullptr, nullptr, nullptr,
                                              o_e2, DD, FF, 64ull * FF, 0);
    }

    // 3) attention pooling over e0
    k_attn_pool<<<BB, 256, 0, stream>>>(e0, w_attn, b_attn, lens, o_sent);

    // 4) bf16 copies of e1 (zero-padded in time) and e2
    {
        int n1 = BB * LP * DD;
        k_pad_e1<<<(n1 + 255) / 256, 256, 0, stream>>>(o_e1, E1p);
        int n2 = MM * DD;
        k_cvt_bf16<<<(n2 / 4 + 255) / 256, 256, 0, stream>>>(o_e2, E2b, n2);
    }

    // 5) verb conv (3-tap) = 3 accumulating WMMA GEMMs over padded e1.
    //    A's 64-row batches are 66*D elements apart (TDM 3D z-stride).
    {
        dim3 grid(MM / 128, CCV / 128);
        for (int t = 0; t < 3; ++t) {
            int flags = (t == 0 ? 0 : 1) | (t == 2 ? 2 : 0);
            k_gemm_wmma<<<grid, 256, 0, stream>>>(E1p + (size_t)t * DD,
                                                  Kvt + (size_t)t * CCV * DD,
                                                  (t == 0) ? bv : nullptr,
                                                  g_v, beta_v, mean_v, var_v,
                                                  LVrb, CCV, DD, (unsigned long long)LP * DD,
                                                  flags);
        }
    }

    // 6) noun conv (1x1) + BN + sigmoid: (8192 x 1024) @ (1024 x 1024)
    {
        dim3 grid(MM / 128, CCN / 128);
        k_gemm_wmma<<<grid, 256, 0, stream>>>(E2b, Knt, bn_b,
                                              g_n, beta_n, mean_n, var_n,
                                              LNun, CCN, DD, 64ull * DD, 2);
    }

    // 7) per-sample top-k mean over time
    {
        int nv = BB * CCV;
        k_topk_mean<<<(nv + 255) / 256, 256, 0, stream>>>(LVrb, lens, o_ilv, CCV);
        int nn = BB * CCN;
        k_topk_mean<<<(nn + 255) / 256, 256, 0, stream>>>(LNun, lens, o_iln, CCN);
    }
}
